// CrossAttention_Video_time_aware_11519102288121
// MI455X (gfx1250) — matrix operand order repaired and checked by reading, not compiled, not run on hardware
//
#include <hip/hip_runtime.h>
#include <hip/hip_bf16.h>

#define B_     4
#define N_     4096
#define QD_    320
#define M_     1024
#define VD_    1024
#define HEADS_ 8
#define DH_    64
#define INNER_ 512
#define W__    128
#define H__    32

typedef __attribute__((ext_vector_type(16))) __bf16         v16bf;
typedef __attribute__((ext_vector_type(16))) unsigned short v16u;
typedef __attribute__((ext_vector_type(8)))  unsigned short v8u;   // 16B
typedef __attribute__((ext_vector_type(8)))  float          v8f;

// ---------- helpers ----------------------------------------------------------

static __device__ __forceinline__ unsigned short f2bf_bits(float f) {
  unsigned int u = __float_as_uint(f);
  u += 0x7fffu + ((u >> 16) & 1u);          // round-to-nearest-even
  return (unsigned short)(u >> 16);
}

// two aligned 16-byte loads -> one 16-element fragment register block
static __device__ __forceinline__ v16u ld2(const unsigned short* p0,
                                           const unsigned short* p1) {
  union { v16u v; v8u h[2]; } u;
  u.h[0] = *(const v8u*)p0;
  u.h[1] = *(const v8u*)p1;
  return u.v;
}

static __device__ __forceinline__ v8f wmma_bf16(v16u a, v16u b, v8f c) {
  return __builtin_amdgcn_wmma_f32_16x16x32_bf16(
      false, __builtin_bit_cast(v16bf, a),
      false, __builtin_bit_cast(v16bf, b),
      (short)0, c, false, false);
}

// ---------- prep kernels -----------------------------------------------------

__global__ __launch_bounds__(256) void maskj_kernel(
    const float* __restrict__ hint, float* __restrict__ maskj) {
  int row = blockIdx.x;                     // b*M + j
  const float* p = hint + (size_t)row * VD_;
  float s = 0.f;
  for (int c = threadIdx.x; c < VD_; c += 256) s += p[c];
  __shared__ float red[256];
  red[threadIdx.x] = s;
  __syncthreads();
  for (int off = 128; off > 0; off >>= 1) {
    if (threadIdx.x < off) red[threadIdx.x] += red[threadIdx.x + off];
    __syncthreads();
  }
  if (threadIdx.x == 0) maskj[row] = (red[0] == 0.f) ? 0.f : 1.f;
}

__global__ __launch_bounds__(128) void maskw_kernel(
    const float* __restrict__ maskj, float* __restrict__ maskw) {
  int b = blockIdx.x, w = threadIdx.x;      // W_ threads
  float pos = (w + 0.5f) * ((float)M_ / (float)W__) - 0.5f;
  pos = fminf(fmaxf(pos, 0.f), (float)(M_ - 1));
  int i0 = (int)floorf(pos);
  int i1 = min(i0 + 1, M_ - 1);
  float t = pos - (float)i0;
  float v = maskj[b * M_ + i0] * (1.f - t) + maskj[b * M_ + i1] * t;
  maskw[b * W__ + w] = (v > 0.5f) ? 1.f : 0.f;
}

__global__ __launch_bounds__(256) void cvt_bf16_kernel(
    const float* __restrict__ src, unsigned short* __restrict__ dst, int n) {
  int i = blockIdx.x * 256 + threadIdx.x;
  if (i < n) dst[i] = f2bf_bits(src[i]);
}

// xr[b, w*H+h, c] = x[b, h*W+w, c]  (fp32 -> bf16, permuted)
__global__ __launch_bounds__(256) void permute_x_kernel(
    const float* __restrict__ x, unsigned short* __restrict__ xr) {
  int idx = blockIdx.x * 256 + threadIdx.x;
  if (idx >= B_ * N_ * QD_) return;
  int c = idx % QD_;
  int t = idx / QD_;
  int i = t % N_;                           // destination row, (w h) order
  int b = t / N_;
  int h = i & (H__ - 1), w = i >> 5;
  xr[idx] = f2bf_bits(x[((size_t)b * N_ + h * W__ + w) * QD_ + c]);
}

// pack weight W[K x N] (f32) into per-lane B-fragment order (bf16):
// Bp[((kc*tiles_n + nc)*32 + lane)*16 + e] = W[kc*32 + (e&7) + 8*(lane>>4) + 16*(e>>3)][nc*16 + (lane&15)]
__global__ __launch_bounds__(256) void pack_w_kernel(
    const float* __restrict__ W, unsigned short* __restrict__ Bp,
    int K, int Ncols) {
  int idx = blockIdx.x * 256 + threadIdx.x;
  if (idx >= K * Ncols) return;
  int e    = idx & 15;
  int lane = (idx >> 4) & 31;
  int t    = idx >> 9;                      // kc * tiles_n + nc
  int tiles_n = Ncols >> 4;
  int kc = t / tiles_n, nc = t % tiles_n;
  int k = kc * 32 + (e & 7) + 8 * (lane >> 4) + 16 * (e >> 3);
  int n = nc * 16 + (lane & 15);
  Bp[idx] = f2bf_bits(W[(size_t)k * Ncols + n]);
}

// ---------- 2x2-blocked GEMM core (32x32 output per wave) --------------------
// acc[0]=(r0,c0) acc[1]=(r0,c1) acc[2]=(r1,c0) acc[3]=(r1,c1)

static __device__ __forceinline__ void tile_gemm_2x2(
    const unsigned short* __restrict__ A, int K, int tiles_n16,
    const unsigned short* __restrict__ Bp, int tm, int tn, int lane,
    v8f acc[4]) {
  int mr = lane & 15, half = lane >> 4;
#pragma unroll
  for (int i = 0; i < 4; ++i)
#pragma unroll
    for (int r = 0; r < 8; ++r) acc[i][r] = 0.f;
  const unsigned short* arow0 = A + (size_t)(tm * 32 + mr) * K + 8 * half;
  const unsigned short* arow1 = arow0 + (size_t)16 * K;
  const unsigned short* bp0 = Bp + ((size_t)(tn * 2) * 32 + lane) * 16;
  const unsigned short* bp1 = bp0 + 32 * 16;
  size_t bstep = (size_t)tiles_n16 * 32 * 16;
  for (int k0 = 0; k0 < K; k0 += 32) {
    v16u a0 = ld2(arow0 + k0, arow0 + k0 + 16);
    v16u a1 = ld2(arow1 + k0, arow1 + k0 + 16);
    v16u b0 = ld2(bp0, bp0 + 8);
    v16u b1 = ld2(bp1, bp1 + 8);
    bp0 += bstep;
    bp1 += bstep;
    acc[0] = wmma_bf16(a0, b0, acc[0]);
    acc[1] = wmma_bf16(a0, b1, acc[1]);
    acc[2] = wmma_bf16(a1, b0, acc[2]);
    acc[3] = wmma_bf16(a1, b1, acc[3]);
  }
}

__global__ __launch_bounds__(256) void gemm_bf16_kernel(
    const unsigned short* __restrict__ A, const unsigned short* __restrict__ Bp,
    unsigned short* __restrict__ C, int Mrows, int K, int Ncols) {
  int lane = threadIdx.x & 31, wave = threadIdx.x >> 5;
  int tiles_n = Ncols >> 5;                 // 32-wide tiles
  int tile = blockIdx.x * 8 + wave;
  int tm = tile / tiles_n, tn = tile % tiles_n;
  if (tm * 32 >= Mrows) return;             // wave-uniform guard
  v8f acc[4];
  tile_gemm_2x2(A, K, Ncols >> 4, Bp, tm, tn, lane, acc);
  int n = lane & 15, g = lane >> 4;
#pragma unroll
  for (int i = 0; i < 2; ++i)
#pragma unroll
    for (int j = 0; j < 2; ++j)
#pragma unroll
      for (int r = 0; r < 8; ++r) {
        int row = tm * 32 + i * 16 + r + 8 * g;
        int col = tn * 32 + j * 16 + n;
        C[(size_t)row * Ncols + col] = f2bf_bits(acc[i * 2 + j][r]);
      }
}

// V projection with transposed store: Vt[(b*INNER + col)*M + key]
__global__ __launch_bounds__(256) void gemm_bf16_vt_kernel(
    const unsigned short* __restrict__ A, const unsigned short* __restrict__ Bp,
    unsigned short* __restrict__ Vt, int Mrows, int K, int Ncols) {
  int lane = threadIdx.x & 31, wave = threadIdx.x >> 5;
  int tiles_n = Ncols >> 5;
  int tile = blockIdx.x * 8 + wave;
  int tm = tile / tiles_n, tn = tile % tiles_n;
  if (tm * 32 >= Mrows) return;
  v8f acc[4];
  tile_gemm_2x2(A, K, Ncols >> 4, Bp, tm, tn, lane, acc);
  int n = lane & 15, g = lane >> 4;
#pragma unroll
  for (int i = 0; i < 2; ++i)
#pragma unroll
    for (int j = 0; j < 2; ++j) {
      int col = tn * 32 + j * 16 + n;       // h*DH + d
#pragma unroll
      for (int r = 0; r < 8; ++r) {
        int row = tm * 32 + i * 16 + r + 8 * g;   // b*M + key
        int b = row >> 10;                  // M_ = 1024
        int key = row & (M_ - 1);
        Vt[((size_t)b * INNER_ + col) * M_ + key] = f2bf_bits(acc[i * 2 + j][r]);
      }
    }
}

// ---------- flash attention (per-wave 16-query tile) -------------------------

__global__ __launch_bounds__(256) void attn_kernel(
    const unsigned short* __restrict__ Qb, const unsigned short* __restrict__ Kb,
    const unsigned short* __restrict__ Vt, const float* __restrict__ maskj,
    const float* __restrict__ maskw, unsigned short* __restrict__ Ob) {
  const float scale2 = 0.125f * 1.44269504f;  // DH^-0.5 * log2(e)
  int lane = threadIdx.x & 31, wave = threadIdx.x >> 5;
  int b = blockIdx.z, h = blockIdx.y;
  int qrow0 = (blockIdx.x * 8 + wave) * 16;
  int mr = lane & 15, half = lane >> 4;

  __shared__ unsigned short ldsP[8][16 * 48];   // stride 48: 16B-aligned rows

  // Q fragments (A-layout): two contiguous 16B runs per 32-wide K chunk
  const unsigned short* qbase =
      Qb + ((size_t)b * N_ + qrow0 + mr) * INNER_ + h * DH_ + 8 * half;
  v16u qa0 = ld2(qbase, qbase + 16);
  v16u qa1 = ld2(qbase + 32, qbase + 48);

  float mi = maskw[b * W__ + (qrow0 >> 5)];     // uniform per 16-query tile

  v8f o0, o1, o2, o3;
  float rmax[8], rsum[8];
#pragma unroll
  for (int r = 0; r < 8; ++r) {
    o0[r] = o1[r] = o2[r] = o3[r] = 0.f;
    rmax[r] = -3.0e38f;
    rsum[r] = 0.f;
  }

  const size_t kvb = (size_t)b * M_;
  const unsigned short* vtb = Vt + ((size_t)b * INNER_ + h * DH_) * M_;

  for (int kb = 0; kb < M_; kb += 32) {
    // prefetch next chunk's K row and Vt run (global_prefetch_b8)
    if (kb + 32 < M_) {
      __builtin_prefetch(Kb + (kvb + kb + 32 + mr) * INNER_ + h * DH_, 0, 3);
      __builtin_prefetch(vtb + (size_t)mr * M_ + kb + 32, 0, 3);
    }

    // ---- S = Q K^T for 32 keys (two 16x16 tiles, K=64 = 2 WMMAs each) ----
    v8f s0, s1;
#pragma unroll
    for (int r = 0; r < 8; ++r) { s0[r] = 0.f; s1[r] = 0.f; }
#pragma unroll
    for (int x = 0; x < 2; ++x) {
      const unsigned short* krow =
          Kb + (kvb + kb + x * 16 + mr) * INNER_ + h * DH_ + 8 * half;
      v16u kv0 = ld2(krow, krow + 16);          // d chunk 0
      v16u kv1 = ld2(krow + 32, krow + 48);     // d chunk 1
      if (x == 0) {
        s0 = wmma_bf16(qa0, kv0, s0);
        s0 = wmma_bf16(qa1, kv1, s0);
      } else {
        s1 = wmma_bf16(qa0, kv0, s1);
        s1 = wmma_bf16(qa1, kv1, s1);
      }
    }

    // ---- scale + mask + online softmax (log2 domain) ----
    float mj0 = maskj[kvb + kb + mr];
    float mj1 = maskj[kvb + kb + 16 + mr];
    float add0 = (mi * mj0 == 0.f) ? -1.0e9f : 0.f;
    float add1 = (mi * mj1 == 0.f) ? -1.0e9f : 0.f;

    float pm[8];
#pragma unroll
    for (int r = 0; r < 8; ++r) {
      s0[r] = s0[r] * scale2 + add0;
      s1[r] = s1[r] * scale2 + add1;
      float v = fmaxf(s0[r], s1[r]);
#pragma unroll
      for (int off = 8; off >= 1; off >>= 1)
        v = fmaxf(v, __shfl_xor(v, off, 16));
      pm[r] = v;
    }
#pragma unroll
    for (int r = 0; r < 8; ++r) {
      float nm = fmaxf(rmax[r], pm[r]);
      float corr = exp2f(rmax[r] - nm);
      rmax[r] = nm;
      float p0 = exp2f(s0[r] - nm);
      float p1 = exp2f(s1[r] - nm);
      float ps = p0 + p1;
#pragma unroll
      for (int off = 8; off >= 1; off >>= 1) ps += __shfl_xor(ps, off, 16);
      rsum[r] = rsum[r] * corr + ps;
      o0[r] *= corr; o1[r] *= corr; o2[r] *= corr; o3[r] *= corr;
      int m = r + 8 * half;
      ldsP[wave][m * 48 + mr]      = f2bf_bits(p0);
      ldsP[wave][m * 48 + 16 + mr] = f2bf_bits(p1);
    }

    // ---- re-read P in A-fragment layout (two aligned ds_load_b128) ----
    const unsigned short* prow = &ldsP[wave][mr * 48 + 8 * half];
    v16u pa = ld2(prow, prow + 16);

    // ---- O += P V (4 column tiles of 16, K=32), Vt is key-contiguous ----
#pragma unroll
    for (int t = 0; t < 4; ++t) {
      const unsigned short* vrow =
          vtb + (size_t)(t * 16 + mr) * M_ + kb + 8 * half;
      v16u vv = ld2(vrow, vrow + 16);
      if (t == 0)      o0 = wmma_bf16(pa, vv, o0);
      else if (t == 1) o1 = wmma_bf16(pa, vv, o1);
      else if (t == 2) o2 = wmma_bf16(pa, vv, o2);
      else             o3 = wmma_bf16(pa, vv, o3);
    }
  }

  // ---- normalize + store bf16 ----
#pragma unroll
  for (int r = 0; r < 8; ++r) {
    float inv = 1.f / rsum[r];
    int m = r + 8 * half;
    size_t base = ((size_t)b * N_ + qrow0 + m) * INNER_ + h * DH_;
    Ob[base + 0  + mr] = f2bf_bits(o0[r] * inv);
    Ob[base + 16 + mr] = f2bf_bits(o1[r] * inv);
    Ob[base + 32 + mr] = f2bf_bits(o2[r] * inv);
    Ob[base + 48 + mr] = f2bf_bits(o3[r] * inv);
  }
}

// ---------- output projection: f32 out, +bias, inverse permute on store ------

__global__ __launch_bounds__(256) void gemm_out_kernel(
    const unsigned short* __restrict__ A, const unsigned short* __restrict__ Bp,
    const float* __restrict__ bo, float* __restrict__ out) {
  int lane = threadIdx.x & 31, wave = threadIdx.x >> 5;
  const int tiles_n = QD_ / 32;             // 10
  int tile = blockIdx.x * 8 + wave;
  int tm = tile / tiles_n, tn = tile % tiles_n;
  if (tm * 32 >= B_ * N_) return;
  v8f acc[4];
  tile_gemm_2x2(A, INNER_, QD_ / 16, Bp, tm, tn, lane, acc);
  int n = lane & 15, g = lane >> 4;
#pragma unroll
  for (int j = 0; j < 2; ++j) {
    int col = tn * 32 + j * 16 + n;
    float bias = bo[col];
#pragma unroll
    for (int i = 0; i < 2; ++i)
#pragma unroll
      for (int r = 0; r < 8; ++r) {
        int row = tm * 32 + i * 16 + r + 8 * g;   // b*N + i, (w h) order
        int b = row / N_;
        int ii = row % N_;
        int h = ii & (H__ - 1), w = ii >> 5;      // dest row = h*W + w
        out[((size_t)b * N_ + h * W__ + w) * QD_ + col] =
            acc[i * 2 + j][r] + bias;
      }
  }
}

// ---------- host launcher ----------------------------------------------------

extern "C" void kernel_launch(void* const* d_in, const int* in_sizes, int n_in,
                              void* d_out, int out_size, void* d_ws,
                              size_t ws_size, hipStream_t stream) {
  const float* x    = (const float*)d_in[0];
  const float* hint = (const float*)d_in[1];
  const float* Wq   = (const float*)d_in[2];
  const float* Wk   = (const float*)d_in[3];
  const float* Wv   = (const float*)d_in[4];
  const float* Wo   = (const float*)d_in[5];
  const float* bo   = (const float*)d_in[6];
  float* out = (float*)d_out;

  char* ws = (char*)d_ws;
  size_t off = 0;
  auto take = [&](size_t bytes) -> char* {
    char* p = ws + off;
    off += (bytes + 255) & ~(size_t)255;
    return p;
  };

  float* maskj = (float*)take((size_t)B_ * M_ * 4);
  float* maskw = (float*)take((size_t)B_ * W__ * 4);
  unsigned short* hintb = (unsigned short*)take((size_t)B_ * M_ * VD_ * 2);
  unsigned short* xrb   = (unsigned short*)take((size_t)B_ * N_ * QD_ * 2);
  unsigned short* wqp   = (unsigned short*)take((size_t)QD_ * INNER_ * 2);
  unsigned short* wkp   = (unsigned short*)take((size_t)VD_ * INNER_ * 2);
  unsigned short* wvp   = (unsigned short*)take((size_t)VD_ * INNER_ * 2);
  unsigned short* wop   = (unsigned short*)take((size_t)INNER_ * QD_ * 2);
  unsigned short* qb    = (unsigned short*)take((size_t)B_ * N_ * INNER_ * 2);
  unsigned short* kb    = (unsigned short*)take((size_t)B_ * M_ * INNER_ * 2);
  unsigned short* vt    = (unsigned short*)take((size_t)B_ * M_ * INNER_ * 2);
  unsigned short* aob   = (unsigned short*)take((size_t)B_ * N_ * INNER_ * 2);

  // masks
  maskj_kernel<<<B_ * M_, 256, 0, stream>>>(hint, maskj);
  maskw_kernel<<<B_, 128, 0, stream>>>(maskj, maskw);

  // fp32 -> bf16 converts / weight fragment packing
  int nh = B_ * M_ * VD_;
  cvt_bf16_kernel<<<(nh + 255) / 256, 256, 0, stream>>>(hint, hintb, nh);
  int nx = B_ * N_ * QD_;
  permute_x_kernel<<<(nx + 255) / 256, 256, 0, stream>>>(x, xrb);
  int nwq = QD_ * INNER_;
  pack_w_kernel<<<(nwq + 255) / 256, 256, 0, stream>>>(Wq, wqp, QD_, INNER_);
  int nwk = VD_ * INNER_;
  pack_w_kernel<<<(nwk + 255) / 256, 256, 0, stream>>>(Wk, wkp, VD_, INNER_);
  pack_w_kernel<<<(nwk + 255) / 256, 256, 0, stream>>>(Wv, wvp, VD_, INNER_);
  int nwo = INNER_ * QD_;
  pack_w_kernel<<<(nwo + 255) / 256, 256, 0, stream>>>(Wo, wop, INNER_, QD_);

  // projections (WMMA bf16, 32x32 tile per wave)
  {
    int tiles = (B_ * N_ / 32) * (INNER_ / 32);   // 8192
    gemm_bf16_kernel<<<tiles / 8, 256, 0, stream>>>(xrb, wqp, qb,
                                                    B_ * N_, QD_, INNER_);
  }
  {
    int tiles = (B_ * M_ / 32) * (INNER_ / 32);   // 2048
    gemm_bf16_kernel<<<tiles / 8, 256, 0, stream>>>(hintb, wkp, kb,
                                                    B_ * M_, VD_, INNER_);
    gemm_bf16_vt_kernel<<<tiles / 8, 256, 0, stream>>>(hintb, wvp, vt,
                                                       B_ * M_, VD_, INNER_);
  }

  // flash attention (WMMA bf16, online softmax)
  {
    dim3 grid(N_ / 16 / 8, HEADS_, B_);           // (32, 8, 4)
    attn_kernel<<<grid, 256, 0, stream>>>(qb, kb, vt, maskj, maskw, aob);
  }

  // output projection + bias + inverse permute
  {
    int tiles = (B_ * N_ / 32) * (QD_ / 32);      // 5120
    gemm_out_kernel<<<tiles / 8, 256, 0, stream>>>(aob, wop, bo, out);
  }
}
